// DecoderLayer_59536836657247
// MI455X (gfx1250) — compile-verified, matrix operand order checked by reading, not run on hardware
//
#include <hip/hip_runtime.h>
#include <hip/hip_bf16.h>

// ---------------------------------------------------------------------------
// CDNA5 (gfx1250, wave32) bf16-WMMA transformer decoder layer.
// GEMMs + attention run on v_wmma_f32_16x16x32_bf16 with fp32 accumulation.
// GEMM A-tiles staged with async global->LDS copies (ASYNCcnt), double-
// buffered LDS software pipeline, global_prefetch on the weight stream.
// ---------------------------------------------------------------------------

typedef __bf16 bf16;
typedef __attribute__((ext_vector_type(8)))  bf16  v8bf;
typedef __attribute__((ext_vector_type(16))) bf16  v16bf;
typedef __attribute__((ext_vector_type(8)))  float v8f;
typedef int v4i_vs __attribute__((vector_size(16)));   // matches builtin param

union AFrag { v16bf v; v8bf h[2]; };

#define WMMA_BF16(a, b, c) \
  __builtin_amdgcn_wmma_f32_16x16x32_bf16(false, (a), false, (b), (short)0, (c), false, false)

#if defined(__has_builtin)
#if __has_builtin(__builtin_amdgcn_global_load_async_to_lds_b128)
#define HAVE_ASYNC_LDS 1
#endif
#endif
#ifndef HAVE_ASYNC_LDS
#define HAVE_ASYNC_LDS 0
#endif

#if HAVE_ASYNC_LDS
// signature (confirmed by round-2 diagnostics): 4 args, global int4* src,
// LDS int4* dst, plus two i32s (offset, cpol).
#define ASYNC_COPY_B128(gsrc, ldst)                                        \
  __builtin_amdgcn_global_load_async_to_lds_b128(                          \
      (__attribute__((address_space(1))) v4i_vs*)(gsrc),                   \
      (__attribute__((address_space(3))) v4i_vs*)(ldst), 0, 0)
#if __has_builtin(__builtin_amdgcn_s_wait_asynccnt)
#define WAIT_ASYNC(n) __builtin_amdgcn_s_wait_asynccnt(n)
#else
#define WAIT_ASYNC(n) asm volatile("s_wait_asynccnt %0" ::"i"(n) : "memory")
#endif
#endif

// ---------------------------------------------------------------------------
// fp32 -> bf16 conversion (grid-stride)
// ---------------------------------------------------------------------------
__global__ __launch_bounds__(256)
void f32_to_bf16_kernel(const float* __restrict__ in, bf16* __restrict__ out, long n) {
  long i = (long)blockIdx.x * blockDim.x + threadIdx.x;
  long stride = (long)gridDim.x * blockDim.x;
  for (; i < n; i += stride) out[i] = (bf16)in[i];
}

// ---------------------------------------------------------------------------
// bf16 GEMM: C[M,N] = A[M,K] @ B[K,N] + bias  (optional ReLU)
// Block = 256 threads = 8 waves. Block tile 128x128; wave tile 32x64
// (2x4 WMMA 16x16x32 tiles). K consumed in steps of 32 through a
// double-buffered LDS stage; A tile copied global->LDS asynchronously,
// B tile transposed through registers (Bt[n][k]). LDS row stride padded to
// 40 halves (80B) so fragment chunks stay 16B aligned and banks spread.
// ---------------------------------------------------------------------------
template <bool RELU>
__global__ __launch_bounds__(256)
void gemm_bf16_kernel(const bf16* __restrict__ A, const bf16* __restrict__ Bw,
                      const float* __restrict__ bias,
                      float* __restrict__ Cf, bf16* __restrict__ Cb,
                      int M, int N, int K) {
  constexpr int LDT = 40;
  __shared__ bf16 As[2][128 * LDT];
  __shared__ bf16 Bts[2][128 * LDT];

  const int tid  = threadIdx.x;
  const int lane = tid & 31;
  const int wave = tid >> 5;
  const int wm   = wave >> 1;          // 0..3 : 32-row group
  const int wn   = wave & 1;           // 0..1 : 64-col group
  const int m0   = blockIdx.y * 128;
  const int n0   = blockIdx.x * 128;

  const v8f vz = {0.f, 0.f, 0.f, 0.f, 0.f, 0.f, 0.f, 0.f};
  v8f acc[2][4];
#pragma unroll
  for (int i = 0; i < 2; ++i)
#pragma unroll
    for (int j = 0; j < 4; ++j) acc[i][j] = vz;

  // staging coordinates
  const int a_row = tid >> 1;            // 0..127
  const int a_cb  = (tid & 1) * 16;      // 0 / 16
  const int b_k   = tid & 31;            // 0..31
  const int b_n16 = (tid >> 5) * 16;     // 0..112

  // fragment coordinates (ISA 16-bit A/B layout)
  const int kb = (lane >> 4) * 8;        // K sub-base: 0 / 8
  const int fl = lane & 15;              // M (or N) index within tile

  const bf16* Ag = A  + (size_t)(m0 + a_row) * K + a_cb;
  const bf16* Bg = Bw + (size_t)b_k * N + n0 + b_n16;
  const int KT = K >> 5;                 // number of 32-wide K tiles

  // ---- prologue: start tile 0 --------------------------------------------
#if HAVE_ASYNC_LDS
  ASYNC_COPY_B128(Ag,     &As[0][a_row * LDT + a_cb]);
  ASYNC_COPY_B128(Ag + 8, &As[0][a_row * LDT + a_cb + 8]);
#else
  v8bf ar0 = *(const v8bf*)(Ag);
  v8bf ar1 = *(const v8bf*)(Ag + 8);
#endif
  v8bf br0 = *(const v8bf*)(Bg);
  v8bf br1 = *(const v8bf*)(Bg + 8);

  for (int kt = 0; kt < KT; ++kt) {
    const int cur = kt & 1;
    const int nxt = cur ^ 1;

    // stage current B tile (transposed) from prefetched registers
    {
      bf16* bt = &Bts[cur][b_n16 * LDT + b_k];
#pragma unroll
      for (int j = 0; j < 8; ++j) bt[j * LDT] = br0[j];
#pragma unroll
      for (int j = 0; j < 8; ++j) bt[(8 + j) * LDT] = br1[j];
    }
#if !HAVE_ASYNC_LDS
    *(v8bf*)(&As[cur][a_row * LDT + a_cb])     = ar0;
    *(v8bf*)(&As[cur][a_row * LDT + a_cb + 8]) = ar1;
#endif

    // kick off tile kt+1 while tile kt is consumed
    if (kt + 1 < KT) {
      const bf16* ga = Ag + (kt + 1) * 32;
      const bf16* gb = Bg + (size_t)(kt + 1) * 32 * N;
#if HAVE_ASYNC_LDS
      ASYNC_COPY_B128(ga,     &As[nxt][a_row * LDT + a_cb]);
      ASYNC_COPY_B128(ga + 8, &As[nxt][a_row * LDT + a_cb + 8]);
#else
      ar0 = *(const v8bf*)(ga);
      ar1 = *(const v8bf*)(ga + 8);
#endif
      br0 = *(const v8bf*)(gb);
      br1 = *(const v8bf*)(gb + 8);
      if (kt + 2 < KT) __builtin_prefetch(Bg + (size_t)(kt + 2) * 32 * N, 0, 1);
#if HAVE_ASYNC_LDS
      WAIT_ASYNC(2);   // tile kt complete; tile kt+1 may stay in flight
#endif
    } else {
#if HAVE_ASYNC_LDS
      WAIT_ASYNC(0);
#endif
    }
    __syncthreads();   // tile kt visible to all waves

    // ---- compute: 2x4 WMMA on the current buffers ------------------------
    AFrag af[2], bfr[4];
#pragma unroll
    for (int i = 0; i < 2; ++i) {
      const int r = wm * 32 + i * 16 + fl;
      af[i].h[0] = *(const v8bf*)(&As[cur][r * LDT + kb]);
      af[i].h[1] = *(const v8bf*)(&As[cur][r * LDT + kb + 16]);
    }
#pragma unroll
    for (int j = 0; j < 4; ++j) {
      const int c = wn * 64 + j * 16 + fl;
      bfr[j].h[0] = *(const v8bf*)(&Bts[cur][c * LDT + kb]);
      bfr[j].h[1] = *(const v8bf*)(&Bts[cur][c * LDT + kb + 16]);
    }
#pragma unroll
    for (int i = 0; i < 2; ++i)
#pragma unroll
      for (int j = 0; j < 4; ++j) acc[i][j] = WMMA_BF16(af[i].v, bfr[j].v, acc[i][j]);
    __syncthreads();   // all waves done reading before buffer reuse
  }

  // ---- epilogue: C layout — lane half h, VGPR r -> row 8h+r, col lane%16 --
  const int half = lane >> 4;
#pragma unroll
  for (int j = 0; j < 4; ++j) {
    const int col = n0 + wn * 64 + j * 16 + fl;
    const float bv = bias ? bias[col] : 0.f;
#pragma unroll
    for (int i = 0; i < 2; ++i) {
#pragma unroll
      for (int r = 0; r < 8; ++r) {
        const int row = m0 + wm * 32 + i * 16 + half * 8 + r;
        float v = acc[i][j][r] + bv;
        if (RELU) v = v > 0.f ? v : 0.f;
        const size_t off = (size_t)row * N + col;
        if (Cf) Cf[off] = v;
        if (Cb) Cb[off] = (bf16)v;
      }
    }
  }
}

// ---------------------------------------------------------------------------
// Flash attention (64 queries per block, hd = 64). One block per (b, h).
// 4 waves; wave w owns query rows 16w..16w+15. K/V streamed in 64-key chunks,
// online softmax in fp32, QK^T and P@V via bf16 WMMA.
// ---------------------------------------------------------------------------
__global__ __launch_bounds__(128)
void attention_kernel(const bf16* __restrict__ Q, long q_bs, int q_hs, int q_rs,
                      const bf16* __restrict__ Kp, const bf16* __restrict__ Vp,
                      long kv_bs, int kv_hs, int kv_rs,
                      const float* __restrict__ mask,            // [B, 64, Skv]
                      bf16* __restrict__ Out, long o_bs, int o_hs, int o_rs,
                      int Skv, float scale, int H) {
  constexpr int LD = 72;   // padded half-stride (144B rows; 16B-aligned chunks)
  __shared__ bf16 Qs[64 * LD];
  __shared__ bf16 Ks[64 * LD];        // K chunk, [key][d]  == Bt for QK^T
  __shared__ bf16 Vts[64 * LD];       // V chunk transposed, [d][key] == Bt for P@V
  __shared__ bf16 Ps[4 * 16 * LD];    // per-wave P strip (16x64)

  const int b    = blockIdx.x / H;
  const int h    = blockIdx.x % H;
  const int tid  = threadIdx.x;
  const int lane = tid & 31;
  const int wave = tid >> 5;           // 0..3
  const int half = lane >> 4;
  const int fl   = lane & 15;
  const int kb   = half * 8;

  const bf16* Qb = Q  + (size_t)b * q_bs  + (size_t)h * q_hs;
  const bf16* Kb = Kp + (size_t)b * kv_bs + (size_t)h * kv_hs;
  const bf16* Vb = Vp + (size_t)b * kv_bs + (size_t)h * kv_hs;
  const float* Mb = mask + (size_t)b * 64 * Skv;

  {   // stage Q (64x64)
    const int row = tid >> 1;
    const int cb  = (tid & 1) * 32;
    const bf16* g = Qb + (size_t)row * q_rs + cb;
#pragma unroll
    for (int c = 0; c < 4; ++c)
      *(v8bf*)(&Qs[row * LD + cb + c * 8]) = *(const v8bf*)(g + c * 8);
  }
  __syncthreads();

  AFrag aq[2];   // Q fragments held in registers for the whole kernel
#pragma unroll
  for (int ks = 0; ks < 2; ++ks) {
    const int r = wave * 16 + fl;
    aq[ks].h[0] = *(const v8bf*)(&Qs[r * LD + ks * 32 + kb]);
    aq[ks].h[1] = *(const v8bf*)(&Qs[r * LD + ks * 32 + kb + 16]);
  }

  const v8f vz = {0.f, 0.f, 0.f, 0.f, 0.f, 0.f, 0.f, 0.f};
  float m[8], l[8];
  v8f oacc[4];
#pragma unroll
  for (int r = 0; r < 8; ++r) { m[r] = -1e30f; l[r] = 0.f; }
#pragma unroll
  for (int t = 0; t < 4; ++t) oacc[t] = vz;

  const int nchunks = Skv >> 6;
  for (int c = 0; c < nchunks; ++c) {
    __syncthreads();
    {   // stage K chunk [key][d]
      const int row = tid >> 1;
      const int cb  = (tid & 1) * 32;
      const bf16* g = Kb + (size_t)(c * 64 + row) * kv_rs + cb;
#pragma unroll
      for (int cc = 0; cc < 4; ++cc)
        *(v8bf*)(&Ks[row * LD + cb + cc * 8]) = *(const v8bf*)(g + cc * 8);
      if (c + 1 < nchunks)
        __builtin_prefetch(Kb + (size_t)((c + 1) * 64 + row) * kv_rs + cb, 0, 1);
    }
    {   // stage V chunk transposed [d][key]
      const int j  = tid >> 1;
      const int db = (tid & 1) * 32;
      const bf16* g = Vb + (size_t)(c * 64 + j) * kv_rs + db;
      v8bf t0 = *(const v8bf*)(g);
      v8bf t1 = *(const v8bf*)(g + 8);
      v8bf t2 = *(const v8bf*)(g + 16);
      v8bf t3 = *(const v8bf*)(g + 24);
#pragma unroll
      for (int i = 0; i < 8; ++i) {
        Vts[(db + i)      * LD + j] = t0[i];
        Vts[(db + 8 + i)  * LD + j] = t1[i];
        Vts[(db + 16 + i) * LD + j] = t2[i];
        Vts[(db + 24 + i) * LD + j] = t3[i];
      }
    }
    __syncthreads();

    // ---- S chunk = scale * Q K^T + mask -------------------------------
    float sc[4][8];
#pragma unroll
    for (int t = 0; t < 4; ++t) {
      v8f s = vz;
#pragma unroll
      for (int ks = 0; ks < 2; ++ks) {
        AFrag bk;
        const int jn = t * 16 + fl;
        bk.h[0] = *(const v8bf*)(&Ks[jn * LD + ks * 32 + kb]);
        bk.h[1] = *(const v8bf*)(&Ks[jn * LD + ks * 32 + kb + 16]);
        s = WMMA_BF16(aq[ks].v, bk.v, s);
      }
      const int j = c * 64 + t * 16 + fl;
#pragma unroll
      for (int r = 0; r < 8; ++r) {
        const int row = wave * 16 + half * 8 + r;
        sc[t][r] = s[r] * scale + Mb[(size_t)row * Skv + j];
      }
    }

    // ---- online softmax (rows live in 16-lane halves) -----------------
    float pm[8];
#pragma unroll
    for (int r = 0; r < 8; ++r) {
      float v = sc[0][r];
      v = fmaxf(v, sc[1][r]); v = fmaxf(v, sc[2][r]); v = fmaxf(v, sc[3][r]);
      pm[r] = v;
    }
#pragma unroll
    for (int x = 1; x < 16; x <<= 1)
#pragma unroll
      for (int r = 0; r < 8; ++r) pm[r] = fmaxf(pm[r], __shfl_xor(pm[r], x, 32));

    float al[8], rs[8];
#pragma unroll
    for (int r = 0; r < 8; ++r) {
      const float nm = fmaxf(m[r], pm[r]);
      al[r] = __expf(m[r] - nm);
      m[r]  = nm;
      rs[r] = 0.f;
    }
#pragma unroll
    for (int t = 0; t < 4; ++t)
#pragma unroll
      for (int r = 0; r < 8; ++r) {
        const float p = __expf(sc[t][r] - m[r]);
        sc[t][r] = p;
        rs[r] += p;
      }
#pragma unroll
    for (int x = 1; x < 16; x <<= 1)
#pragma unroll
      for (int r = 0; r < 8; ++r) rs[r] += __shfl_xor(rs[r], x, 32);
#pragma unroll
    for (int r = 0; r < 8; ++r) l[r] = l[r] * al[r] + rs[r];
#pragma unroll
    for (int t = 0; t < 4; ++t)
#pragma unroll
      for (int r = 0; r < 8; ++r) oacc[t][r] *= al[r];

    // ---- P -> LDS (re-fragment as A operand), then O += P V -----------
    bf16* Pw = &Ps[wave * 16 * LD];
#pragma unroll
    for (int t = 0; t < 4; ++t)
#pragma unroll
      for (int r = 0; r < 8; ++r)
        Pw[(half * 8 + r) * LD + t * 16 + fl] = (bf16)sc[t][r];
    __syncthreads();

    AFrag ap[2];
#pragma unroll
    for (int ks = 0; ks < 2; ++ks) {
      ap[ks].h[0] = *(const v8bf*)(&Pw[fl * LD + ks * 32 + kb]);
      ap[ks].h[1] = *(const v8bf*)(&Pw[fl * LD + ks * 32 + kb + 16]);
    }
#pragma unroll
    for (int t = 0; t < 4; ++t) {
#pragma unroll
      for (int ks = 0; ks < 2; ++ks) {
        AFrag bv;
        const int d = t * 16 + fl;
        bv.h[0] = *(const v8bf*)(&Vts[d * LD + ks * 32 + kb]);
        bv.h[1] = *(const v8bf*)(&Vts[d * LD + ks * 32 + kb + 16]);
        oacc[t] = WMMA_BF16(ap[ks].v, bv.v, oacc[t]);
      }
    }
  }

  // ---- normalize and write O (bf16, feeds the next GEMM) ---------------
#pragma unroll
  for (int t = 0; t < 4; ++t) {
#pragma unroll
    for (int r = 0; r < 8; ++r) {
      const int row = wave * 16 + half * 8 + r;
      const int d   = t * 16 + fl;
      const float v = oacc[t][r] / l[r];
      Out[(size_t)b * o_bs + (size_t)h * o_hs + (size_t)row * o_rs + d] = (bf16)v;
    }
  }
}

// ---------------------------------------------------------------------------
// out = LayerNorm(src + resid) * gamma + beta.  One row (D=512) per block.
// ---------------------------------------------------------------------------
__global__ __launch_bounds__(256)
void add_ln_kernel(const float* __restrict__ src, const float* __restrict__ resid,
                   const float* __restrict__ gamma, const float* __restrict__ beta,
                   float* __restrict__ outf, bf16* __restrict__ outb, int Dd) {
  __shared__ float red[16];
  const int row = blockIdx.x;
  const int tid = threadIdx.x;
  const int i0  = tid * 2;
  const float* s = src   + (size_t)row * Dd;
  const float* q = resid + (size_t)row * Dd;
  const float v0 = s[i0] + q[i0];
  const float v1 = s[i0 + 1] + q[i0 + 1];
  float sum = v0 + v1;
  float sq  = v0 * v0 + v1 * v1;
#pragma unroll
  for (int x = 1; x < 32; x <<= 1) {
    sum += __shfl_xor(sum, x, 32);
    sq  += __shfl_xor(sq,  x, 32);
  }
  const int wave = tid >> 5, lane = tid & 31;
  if (lane == 0) { red[wave] = sum; red[8 + wave] = sq; }
  __syncthreads();
  float fs = 0.f, fq = 0.f;
#pragma unroll
  for (int i = 0; i < 8; ++i) { fs += red[i]; fq += red[8 + i]; }
  const float mean = fs / Dd;
  const float var  = fq / Dd - mean * mean;
  const float rstd = rsqrtf(var + 1e-5f);
  const float o0 = gamma[i0]     * (v0 - mean) * rstd + beta[i0];
  const float o1 = gamma[i0 + 1] * (v1 - mean) * rstd + beta[i0 + 1];
  outf[(size_t)row * Dd + i0]     = o0;
  outf[(size_t)row * Dd + i0 + 1] = o1;
  if (outb) {
    outb[(size_t)row * Dd + i0]     = (bf16)o0;
    outb[(size_t)row * Dd + i0 + 1] = (bf16)o1;
  }
}

// ---------------------------------------------------------------------------
// Host-side orchestration
// ---------------------------------------------------------------------------
extern "C" void kernel_launch(void* const* d_in, const int* in_sizes, int n_in,
                              void* d_out, int out_size, void* d_ws, size_t ws_size,
                              hipStream_t stream) {
  const int B = 256, Sd = 64, Se = 512, D = 512, H = 8, FF = 2048;
  const int Md = B * Sd;       // 16384 decoder tokens
  const int Me = B * Se;       // 131072 encoder tokens

  const float* x     = (const float*)d_in[0];
  const float* y     = (const float*)d_in[1];
  const float* mask  = (const float*)d_in[2];
  const float* mask2 = (const float*)d_in[3];
  const float* qkv_w = (const float*)d_in[4];  const float* qkv_b = (const float*)d_in[5];
  const float* sa_w  = (const float*)d_in[6];  const float* sa_b  = (const float*)d_in[7];
  const float* q_w   = (const float*)d_in[8];  const float* q_b   = (const float*)d_in[9];
  const float* kv_w  = (const float*)d_in[10]; const float* kv_b  = (const float*)d_in[11];
  const float* ca_w  = (const float*)d_in[12]; const float* ca_b  = (const float*)d_in[13];
  const float* l1_w  = (const float*)d_in[14]; const float* l1_b  = (const float*)d_in[15];
  const float* l2_w  = (const float*)d_in[16]; const float* l2_b  = (const float*)d_in[17];
  const float* g1 = (const float*)d_in[18]; const float* b1 = (const float*)d_in[19];
  const float* g2 = (const float*)d_in[20]; const float* b2 = (const float*)d_in[21];
  const float* g3 = (const float*)d_in[22]; const float* b3 = (const float*)d_in[23];

  // workspace layout (~797 MB total)
  char* wp = (char*)d_ws;
  auto alloc = [&](size_t bytes) -> char* {
    char* p = wp;
    wp += (bytes + 255) & ~(size_t)255;
    return p;
  };
  bf16* yb    = (bf16*)alloc((size_t)Md * D * 2);
  bf16* xb    = (bf16*)alloc((size_t)Me * D * 2);
  bf16* qkvwb = (bf16*)alloc((size_t)D * 3 * D * 2);
  bf16* sawb  = (bf16*)alloc((size_t)D * D * 2);
  bf16* qwb   = (bf16*)alloc((size_t)D * D * 2);
  bf16* kvwb  = (bf16*)alloc((size_t)D * 2 * D * 2);
  bf16* cawb  = (bf16*)alloc((size_t)D * D * 2);
  bf16* l1wb  = (bf16*)alloc((size_t)D * FF * 2);
  bf16* l2wb  = (bf16*)alloc((size_t)FF * D * 2);
  bf16* qkvb  = (bf16*)alloc((size_t)Md * 3 * D * 2);
  bf16* vals  = (bf16*)alloc((size_t)Md * D * 2);
  float* saf  = (float*)alloc((size_t)Md * D * 4);
  float* y1f  = (float*)alloc((size_t)Md * D * 4);
  bf16* y1b   = (bf16*)alloc((size_t)Md * D * 2);
  bf16* kvb   = (bf16*)alloc((size_t)Me * 2 * D * 2);
  bf16* qcb   = (bf16*)alloc((size_t)Md * D * 2);
  bf16* vals2 = (bf16*)alloc((size_t)Md * D * 2);
  float* caf  = (float*)alloc((size_t)Md * D * 4);
  float* y2f  = (float*)alloc((size_t)Md * D * 4);
  bf16* y2b   = (bf16*)alloc((size_t)Md * D * 2);
  bf16* hb    = (bf16*)alloc((size_t)Md * FF * 2);
  float* fff  = (float*)alloc((size_t)Md * D * 4);

  auto cvt = [&](const float* in, bf16* out, long n) {
    int blocks = (int)(((n + 255) / 256) < 4096 ? ((n + 255) / 256) : 4096);
    f32_to_bf16_kernel<<<blocks, 256, 0, stream>>>(in, out, n);
  };
  auto gemm = [&](const bf16* A, const bf16* Bw, const float* bias,
                  float* Cf, bf16* Cb, int M, int N, int K, bool relu) {
    dim3 grid(N / 128, M / 128);
    if (relu) gemm_bf16_kernel<true><<<grid, 256, 0, stream>>>(A, Bw, bias, Cf, Cb, M, N, K);
    else      gemm_bf16_kernel<false><<<grid, 256, 0, stream>>>(A, Bw, bias, Cf, Cb, M, N, K);
  };

  // 0. conversions
  cvt(y, yb, (long)Md * D);
  cvt(x, xb, (long)Me * D);
  cvt(qkv_w, qkvwb, (long)D * 3 * D);
  cvt(sa_w, sawb, (long)D * D);
  cvt(q_w, qwb, (long)D * D);
  cvt(kv_w, kvwb, (long)D * 2 * D);
  cvt(ca_w, cawb, (long)D * D);
  cvt(l1_w, l1wb, (long)D * FF);
  cvt(l2_w, l2wb, (long)FF * D);

  const float scale = 0.125f;  // 1/sqrt(64)

  // 1. self-attention
  gemm(yb, qkvwb, qkv_b, nullptr, qkvb, Md, 3 * D, D, false);
  // qkv head layout: (..., H, 3*hd) -> per head h: q at h*192, k at +64, v at +128
  attention_kernel<<<B * H, 128, 0, stream>>>(
      qkvb,        (long)Sd * 3 * D, 192, 3 * D,
      qkvb + 64,   qkvb + 128, (long)Sd * 3 * D, 192, 3 * D,
      mask,
      vals, (long)Sd * D, 64, D,
      Sd, scale, H);
  gemm(vals, sawb, sa_b, saf, nullptr, Md, D, D, false);
  add_ln_kernel<<<Md, 256, 0, stream>>>(saf, y, g1, b1, y1f, y1b, D);

  // 2. cross-attention
  gemm(xb, kvwb, kv_b, nullptr, kvb, Me, 2 * D, D, false);   // the big one
  gemm(y1b, qwb, q_b, nullptr, qcb, Md, D, D, false);
  // kv head layout: (..., H, 2*hd) -> per head h: k at h*128, v at +64
  attention_kernel<<<B * H, 128, 0, stream>>>(
      qcb,       (long)Sd * D, 64, D,
      kvb,       kvb + 64, (long)Se * 2 * D, 128, 2 * D,
      mask2,
      vals2, (long)Sd * D, 64, D,
      Se, scale, H);
  gemm(vals2, cawb, ca_b, caf, nullptr, Md, D, D, false);
  add_ln_kernel<<<Md, 256, 0, stream>>>(caf, y1f, g2, b2, y2f, y2b, D);

  // 3. FFN
  gemm(y2b, l1wb, l1_b, nullptr, hb, Md, FF, D, true);   // ReLU
  gemm(hb, l2wb, l2_b, fff, nullptr, Md, D, FF, false);
  add_ln_kernel<<<Md, 256, 0, stream>>>(fff, y2f, g3, b3, (float*)d_out, nullptr, D);
}
